// MGCN2_56908316672075
// MI455X (gfx1250) — matrix operand order repaired and checked by reading, not compiled, not run on hardware
//
#include <hip/hip_runtime.h>
#include <hip/hip_bf16.h>
#include <cstdint>

// ---------------------------------------------------------------------------
// Problem constants (from reference)
// ---------------------------------------------------------------------------
#define NN     10000
#define EE     640000
#define F_IN   256
#define F_OUT  64
#define KHOPS  4
#define F_CAT  1280          // (K+1)*F_IN
#define EMBT_COLS 144        // 64 (emb) + 64 (emb_a) + 16 (ones-block for row sums)
#define EMBT_LD   10016      // 10000 padded up to multiple of 32
#define VSUM_LD   144
#define M_TILES   625        // 10000 / 16

typedef __attribute__((ext_vector_type(16))) __bf16 v16bf;
typedef __attribute__((ext_vector_type(8)))  __bf16 v8bf;
typedef __attribute__((ext_vector_type(8)))  float  v8f;

__device__ __forceinline__ v16bf combine16(v8bf lo, v8bf hi) {
    v16bf r;
#pragma unroll
    for (int i = 0; i < 8; ++i) { r[i] = lo[i]; r[i + 8] = hi[i]; }
    return r;
}

// async 16B global -> LDS copy (per active lane), tracked by ASYNCcnt
__device__ __forceinline__ void async_copy_b128(void* lds_ptr, const void* gptr) {
    uint32_t ldsoff = (uint32_t)(uintptr_t)lds_ptr;   // low 32 bits = LDS offset
    asm volatile("global_load_async_to_lds_b128 %0, %1, off"
                 :: "v"(ldsoff), "v"(gptr) : "memory");
}
__device__ __forceinline__ void wait_asynccnt0() {
    asm volatile("s_wait_asynccnt 0x0" ::: "memory");
}

// ---------------------------------------------------------------------------
// Phase 1: symmetric normalization
// ---------------------------------------------------------------------------
__global__ void k_deg_init(float* deg) {
    int n = blockIdx.x * blockDim.x + threadIdx.x;
    if (n < NN) deg[n] = 1.0f;                    // self-loop weight 1
}

__global__ void k_deg_acc(const int* __restrict__ row, const int* __restrict__ col,
                          float* deg) {
    int e = blockIdx.x * blockDim.x + threadIdx.x;
    if (e >= EE) return;
    int r = row[e], c = col[e];
    if (r != c) unsafeAtomicAdd(&deg[r], 1.0f);
}

__global__ void k_dis(float* deg_dis) {
    int n = blockIdx.x * blockDim.x + threadIdx.x;
    if (n >= NN) return;
    float d = deg_dis[n];
    deg_dis[n] = (d > 0.0f) ? rsqrtf(d) : 0.0f;
}

__global__ void k_norm(const int* __restrict__ row, const int* __restrict__ col,
                       const float* __restrict__ dis, float* __restrict__ norm) {
    int e = blockIdx.x * blockDim.x + threadIdx.x;
    if (e >= EE) return;
    int r = row[e], c = col[e];
    norm[e] = (r != c) ? dis[r] * dis[c] : 0.0f;
}

// ---------------------------------------------------------------------------
// Phase 1b: CSR build (in-edges per destination) -> atomic-free propagation
// ---------------------------------------------------------------------------
__global__ void k_cnt_init(int* cnt) {
    int n = blockIdx.x * blockDim.x + threadIdx.x;
    if (n < NN) cnt[n] = 0;
}

__global__ void k_cnt(const int* __restrict__ row, const int* __restrict__ col,
                      int* cnt) {
    int e = blockIdx.x * blockDim.x + threadIdx.x;
    if (e >= EE) return;
    if (row[e] != col[e]) atomicAdd(&cnt[col[e]], 1);
}

// single-block exclusive scan of cnt[0..NN) -> offsets[0..NN], plus pos copy
#define SCAN_T 256
#define SCAN_CHUNK 40        // 256*40 = 10240 >= NN
__global__ __launch_bounds__(SCAN_T) void k_scan(const int* __restrict__ cnt,
                                                 int* __restrict__ offsets,
                                                 int* __restrict__ pos) {
    __shared__ int part[SCAN_T];
    int t = threadIdx.x;
    int base = t * SCAN_CHUNK;
    int s = 0;
    for (int i = 0; i < SCAN_CHUNK; ++i) {
        int idx = base + i;
        if (idx < NN) s += cnt[idx];
    }
    part[t] = s;
    __syncthreads();
    for (int d = 1; d < SCAN_T; d <<= 1) {      // Hillis-Steele inclusive scan
        int v = (t >= d) ? part[t - d] : 0;
        __syncthreads();
        part[t] += v;
        __syncthreads();
    }
    int run = (t == 0) ? 0 : part[t - 1];
    for (int i = 0; i < SCAN_CHUNK; ++i) {
        int idx = base + i;
        if (idx < NN) { offsets[idx] = run; pos[idx] = run; run += cnt[idx]; }
    }
    if (t == SCAN_T - 1) offsets[NN] = part[SCAN_T - 1];
}

__global__ void k_fill(const int* __restrict__ row, const int* __restrict__ col,
                       int* pos, int* __restrict__ eid) {
    int e = blockIdx.x * blockDim.x + threadIdx.x;
    if (e >= EE) return;
    if (row[e] != col[e]) {
        int idx = atomicAdd(&pos[col[e]], 1);
        eid[idx] = e;
    }
}

// deterministic order: insertion-sort each node's in-edge list by edge id
__global__ void k_sortlists(const int* __restrict__ offsets, int* eid) {
    int n = blockIdx.x * blockDim.x + threadIdx.x;
    if (n >= NN) return;
    int s = offsets[n], e = offsets[n + 1];
    for (int i = s + 1; i < e; ++i) {
        int v = eid[i];
        int j = i - 1;
        while (j >= s && eid[j] > v) { eid[j + 1] = eid[j]; --j; }
        eid[j + 1] = v;
    }
}

// ---------------------------------------------------------------------------
// Phase 2: K-hop propagation into xcat [N, 1280] (gather form, no atomics)
// thread = (node, float4 chunk); each wave serves a single node
// ---------------------------------------------------------------------------
__global__ __launch_bounds__(256) void k_copy_hop0(const float* __restrict__ feat,
                                                   float* __restrict__ xcat) {
    int t = blockIdx.x * blockDim.x + threadIdx.x;     // N * 64 float4 chunks
    if (t >= NN * (F_IN / 4)) return;
    int n = t >> 6, c = t & 63;
    const float4 v = reinterpret_cast<const float4*>(feat)[(size_t)n * 64 + c];
    reinterpret_cast<float4*>(xcat)[(size_t)n * (F_CAT / 4) + c] = v;
}

__global__ __launch_bounds__(256) void k_gather(
        const int* __restrict__ offsets, const int* __restrict__ eid,
        const int* __restrict__ row,     const float* __restrict__ norm,
        const float* __restrict__ dis,   float* __restrict__ xcat, int h)
{
    int t = blockIdx.x * blockDim.x + threadIdx.x;
    if (t >= NN * (F_IN / 4)) return;
    int n = t >> 6, c = t & 63;
    const float4* xp = reinterpret_cast<const float4*>(xcat);
    size_t prevCol = (size_t)(h - 1) * (F_IN / 4) + c;

    float s = dis[n];
    float d2 = s * s;                                   // self-loop norm
    float4 v0 = xp[(size_t)n * (F_CAT / 4) + prevCol];
    float4 acc;
    acc.x = d2 * v0.x; acc.y = d2 * v0.y; acc.z = d2 * v0.z; acc.w = d2 * v0.w;

    int e0 = offsets[n], e1 = offsets[n + 1];
    for (int j = e0; j < e1; ++j) {
        int e = eid[j];
        float w = norm[e];                              // broadcast across wave
        int src = row[e];
        float4 v = xp[(size_t)src * (F_CAT / 4) + prevCol];
        acc.x += w * v.x; acc.y += w * v.y; acc.z += w * v.z; acc.w += w * v.w;
    }
    reinterpret_cast<float4*>(xcat)[(size_t)n * (F_CAT / 4) + (size_t)h * 64 + c] = acc;
}

// f32 -> bf16 conversion, 8 elements/thread
__global__ void k_cvt_bf(const float* __restrict__ x, __bf16* __restrict__ xbf,
                         long long count8) {
    long long t = (long long)blockIdx.x * blockDim.x + threadIdx.x;
    if (t >= count8) return;
    const float4* in4 = reinterpret_cast<const float4*>(x);
    float4 a = in4[2 * t], b = in4[2 * t + 1];
    v8bf o;
    o[0] = (__bf16)a.x; o[1] = (__bf16)a.y; o[2] = (__bf16)a.z; o[3] = (__bf16)a.w;
    o[4] = (__bf16)b.x; o[5] = (__bf16)b.y; o[6] = (__bf16)b.z; o[7] = (__bf16)b.w;
    *reinterpret_cast<v8bf*>(xbf + 8 * t) = o;
}

// W [1280,64] f32 row-major  ->  Wt [64,1280] bf16
__global__ void k_cvt_W(const float* __restrict__ W, __bf16* __restrict__ Wt) {
    int t = blockIdx.x * blockDim.x + threadIdx.x;
    if (t >= F_CAT * F_OUT) return;
    int k = t >> 6, n = t & 63;
    Wt[(size_t)n * F_CAT + k] = (__bf16)W[t];
}

// embT [144][10016] bf16 : zero everything; row 128 = 1.0 for node < N
__global__ void k_embT_init(__bf16* __restrict__ embT) {
    int t = blockIdx.x * blockDim.x + threadIdx.x;
    if (t >= EMBT_COLS * EMBT_LD) return;
    int r = t / EMBT_LD, c = t % EMBT_LD;
    embT[t] = (__bf16)((r == 128 && c < NN) ? 1.0f : 0.0f);
}

// ---------------------------------------------------------------------------
// Phase 3: WMMA GEMM  emb = relu(xcat @ W)   [10000,1280]x[1280,64]
// 8 waves/block, one 16x64 block per wave.
// Ping-pong 2x-unrolled K loop: fragment set 0/1 each written once per
// iteration directly by loads (no register rotation copies); prefetch for
// step i+1 issues before the WMMAs of step i. Final prefetch reads the
// 64-element pad appended to Xbf / Wt.
// ---------------------------------------------------------------------------
__device__ __forceinline__ void load_frags(const __bf16* __restrict__ arow,
                                           const __bf16* __restrict__ Wt,
                                           int m, int kh, int kn,
                                           v16bf& A, v16bf B[4]) {
    A = combine16(*(const v8bf*)(arow + kn + kh * 16),
                  *(const v8bf*)(arow + kn + kh * 16 + 8));
#pragma unroll
    for (int t = 0; t < 4; ++t) {
        const __bf16* bp = Wt + (size_t)(t * 16 + m) * F_CAT + kn + kh * 16;
        B[t] = combine16(*(const v8bf*)bp, *(const v8bf*)(bp + 8));
    }
}

__global__ __launch_bounds__(256) void k_gemm_xW(
        const __bf16* __restrict__ Xbf,   // [10000][1280] (+64 pad)
        const __bf16* __restrict__ Wt,    // [64][1280]    (+64 pad)
        float* __restrict__ outF,         // [10000][64]
        __bf16* __restrict__ embT,        // [144][10016]
        int colBase)
{
    int wave = threadIdx.x >> 5;
    int lane = threadIdx.x & 31;
    int tile = blockIdx.x * 8 + wave;
    if (tile >= M_TILES) return;
    int mbase = tile * 16;
    int m = lane & 15, kh = lane >> 4;

    v8f acc[4] = {};
    const __bf16* arow = Xbf + (size_t)(mbase + m) * F_CAT;

    v16bf A0, A1;
    v16bf B0[4], B1[4];
    load_frags(arow, Wt, m, kh, 0, A0, B0);

    for (int k0 = 0; k0 < F_CAT; k0 += 64) {
        load_frags(arow, Wt, m, kh, k0 + 32, A1, B1);
#pragma unroll
        for (int t = 0; t < 4; ++t) {
            acc[t] = __builtin_amdgcn_wmma_f32_16x16x32_bf16(
                false, A0, false, B0[t], (short)0, acc[t], false, false);
        }
        load_frags(arow, Wt, m, kh, k0 + 64, A0, B0);   // last one reads pad
#pragma unroll
        for (int t = 0; t < 4; ++t) {
            acc[t] = __builtin_amdgcn_wmma_f32_16x16x32_bf16(
                false, A1, false, B1[t], (short)0, acc[t], false, false);
        }
    }
#pragma unroll
    for (int t = 0; t < 4; ++t) {
        int n = t * 16 + m;
        v8bf ebf;
#pragma unroll
        for (int r = 0; r < 8; ++r) {
            float v = acc[t][r];
            v = v > 0.0f ? v : 0.0f;                 // relu
            int node = mbase + kh * 8 + r;           // C layout: M = r + 8*kh
            outF[(size_t)node * F_OUT + n] = v;
            ebf[r] = (__bf16)v;
        }
        *(v8bf*)(embT + (size_t)(colBase + n) * EMBT_LD + mbase + kh * 8) = ebf;
    }
}

// ---------------------------------------------------------------------------
// Phase 4: WMMA GEMM  vsum = mask @ [emb | emb_a | ones]  ->  [10000][144]
// 9 waves/block; A tile (16x32 f32) staged via async global->LDS DMA,
// double-buffered so tile i+1 streams in while tile i feeds the WMMAs.
// ---------------------------------------------------------------------------
__global__ __launch_bounds__(288) void k_gemm_mask(
        const float* __restrict__ mask,     // [10000][10000] f32
        const __bf16* __restrict__ embT,    // [144][10016] bf16
        float* __restrict__ vsum)           // [10000][144] f32
{
    __shared__ float At[2][16 * 32];
    int wave = threadIdx.x >> 5;   // 0..8  -> N-tile
    int lane = threadIdx.x & 31;
    int tid  = threadIdx.x;
    int mbase = blockIdx.x * 16;
    int m = lane & 15, kh = lane >> 4;

    const int NSTEP  = (NN + 31) / 32;   // 313
    const int NASYNC = NN / 32;          // 312 steps have a full 32-wide tile

    // async fill mapping: threads 0..127, 16B each (16 rows x 32 f32 = 2KB)
    int frow = tid >> 3, fk = (tid & 7) * 4;

    v8f acc = {};
    const __bf16* brow = embT + (size_t)(wave * 16 + m) * EMBT_LD;

    // prologue: DMA tile 0
    if (tid < 128) {
        async_copy_b128(&At[0][frow * 32 + fk],
                        mask + (size_t)(mbase + frow) * NN + fk);
        wait_asynccnt0();
    }
    __syncthreads();

    for (int i = 0; i < NSTEP; ++i) {
        int k0 = i * 32;
        int buf = i & 1, nbuf = buf ^ 1;

        if (i + 1 < NASYNC) {
            // stream next full tile while this one computes
            if (tid < 128) {
                async_copy_b128(&At[nbuf][frow * 32 + fk],
                                mask + (size_t)(mbase + frow) * NN + (k0 + 32) + fk);
            }
        } else if (i + 1 == NASYNC) {
            // ragged tail tile (k = 9984..10015): bounds-checked sync fill
            if (tid < 256) {
                int flat = tid * 2;
                int r = flat >> 5, kk = flat & 31;
                int k = (k0 + 32) + kk;
                const float* src = mask + (size_t)(mbase + r) * NN + k;
                At[nbuf][flat]     = (k     < NN) ? src[0] : 0.0f;
                At[nbuf][flat + 1] = (k + 1 < NN) ? src[1] : 0.0f;
            }
        }

        // compute from current buffer
        v16bf A;
#pragma unroll
        for (int j = 0; j < 8; ++j) A[j]     = (__bf16)At[buf][m * 32 + kh * 16 + j];
#pragma unroll
        for (int j = 0; j < 8; ++j) A[j + 8] = (__bf16)At[buf][m * 32 + kh * 16 + 8 + j];
        const __bf16* bp = brow + k0 + kh * 16;
        v16bf B = combine16(*(const v8bf*)bp, *(const v8bf*)(bp + 8));
        acc = __builtin_amdgcn_wmma_f32_16x16x32_bf16(
            false, A, false, B, (short)0, acc, false, false);

        if (tid < 128) wait_asynccnt0();   // next tile fully in LDS
        __syncthreads();
    }
#pragma unroll
    for (int r = 0; r < 8; ++r) {
        int node = mbase + kh * 8 + r;
        vsum[(size_t)node * VSUM_LD + wave * 16 + m] = acc[r];
    }
}

// ---------------------------------------------------------------------------
// Phase 5: avg-readout finalize: divide by row-sum, L2-normalize, sigmoid
// ---------------------------------------------------------------------------
__global__ void k_readout(const float* __restrict__ vsum,
                          float* __restrict__ g, float* __restrict__ ga)
{
    int n = blockIdx.x * blockDim.x + threadIdx.x;
    if (n >= NN) return;
    const float* row = vsum + (size_t)n * VSUM_LD;
    float inv_rs = 1.0f / row[128];                 // row sum (ones column)
    for (int which = 0; which < 2; ++which) {
        const float* v = row + which * 64;
        float ss = 0.0f;
        for (int f = 0; f < 64; ++f) { float x = v[f] * inv_rs; ss += x * x; }
        float inv = 1.0f / fmaxf(sqrtf(ss), 1e-12f);
        float* dst = (which ? ga : g) + (size_t)n * 64;
        for (int f = 0; f < 64; ++f) {
            float x = v[f] * inv_rs * inv;
            dst[f] = 1.0f / (1.0f + expf(-x));
        }
    }
}

// ---------------------------------------------------------------------------
// Phase 6: bilinear discriminator: sc = h^T Wb c + b
// ---------------------------------------------------------------------------
__global__ __launch_bounds__(256) void k_disc(
        const float* __restrict__ emb, const float* __restrict__ emba,
        const float* __restrict__ g,   const float* __restrict__ ga,
        const float* __restrict__ Wb,  const float* __restrict__ bptr,
        float* __restrict__ ret, float* __restrict__ reta)
{
    __shared__ float WbL[64 * 64];
    for (int i = threadIdx.x; i < 64 * 64; i += blockDim.x) WbL[i] = Wb[i];
    __syncthreads();
    int n = blockIdx.x * blockDim.x + threadIdx.x;
    if (n >= NN) return;
    float b = bptr[0];
    const float* h  = emb  + (size_t)n * 64;
    const float* hm = emba + (size_t)n * 64;
    const float* c  = g    + (size_t)n * 64;
    const float* cm = ga   + (size_t)n * 64;
    float sc1 = 0, sc2 = 0, sa1 = 0, sa2 = 0;
    for (int d = 0; d < 64; ++d) {
        const float* wr = WbL + d * 64;
        float s = 0, s2 = 0;
        for (int e = 0; e < 64; ++e) { s += wr[e] * c[e]; s2 += wr[e] * cm[e]; }
        float hd = h[d], hmd = hm[d];
        sc1 += hd  * s;    // emb   . Wb . g
        sc2 += hmd * s;    // emb_a . Wb . g
        sa1 += hmd * s2;   // emb_a . Wb . g_a
        sa2 += hd  * s2;   // emb   . Wb . g_a
    }
    ret [n * 2 + 0] = sc1 + b;  ret [n * 2 + 1] = sc2 + b;
    reta[n * 2 + 0] = sa1 + b;  reta[n * 2 + 1] = sa2 + b;
}

// ---------------------------------------------------------------------------
// Host launcher
// ---------------------------------------------------------------------------
static inline size_t alignup(size_t x) { return (x + 255) & ~(size_t)255; }

extern "C" void kernel_launch(void* const* d_in, const int* in_sizes, int n_in,
                              void* d_out, int out_size, void* d_ws, size_t ws_size,
                              hipStream_t stream) {
    const float* feat   = (const float*)d_in[0];
    const float* feat_a = (const float*)d_in[1];
    const int*   eidx   = (const int*)  d_in[2];
    const float* mask   = (const float*)d_in[3];
    const float* W      = (const float*)d_in[4];
    const float* Wb     = (const float*)d_in[5];
    const float* bsc    = (const float*)d_in[6];
    const int* erow = eidx;
    const int* ecol = eidx + EE;

    // workspace carve-up (xbf / Wt get a 64-element bf16 pad for the
    // unconditional last prefetch in k_gemm_xW)
    char* ws = (char*)d_ws;
    size_t off = 0;
    float*  xcat  = (float*) (ws + off); off = alignup(off + (size_t)NN * F_CAT * 4);
    __bf16* xbf   = (__bf16*)(ws + off); off = alignup(off + ((size_t)NN * F_CAT + 64) * 2);
    float*  norm  = (float*) (ws + off); off = alignup(off + (size_t)EE * 4);
    float*  dis   = (float*) (ws + off); off = alignup(off + (size_t)NN * 4);
    __bf16* Wt    = (__bf16*)(ws + off); off = alignup(off + ((size_t)F_OUT * F_CAT + 64) * 2);
    __bf16* embT  = (__bf16*)(ws + off); off = alignup(off + (size_t)EMBT_COLS * EMBT_LD * 2);
    float*  emb_a = (float*) (ws + off); off = alignup(off + (size_t)NN * F_OUT * 4);
    float*  vsum  = (float*) (ws + off); off = alignup(off + (size_t)NN * VSUM_LD * 4);
    float*  g     = (float*) (ws + off); off = alignup(off + (size_t)NN * F_OUT * 4);
    float*  ga    = (float*) (ws + off); off = alignup(off + (size_t)NN * F_OUT * 4);
    int*    csr_off = (int*)(ws + off);  off = alignup(off + (size_t)(NN + 1) * 4);
    int*    csr_pos = (int*)(ws + off);  off = alignup(off + (size_t)NN * 4);
    int*    csr_eid = (int*)(ws + off);  off = alignup(off + (size_t)EE * 4);

    float* out_emb = (float*)d_out;                  // [10000,64]
    float* out_ret = out_emb + (size_t)NN * F_OUT;   // [10000,2]
    float* out_ra  = out_ret + (size_t)NN * 2;       // [10000,2]

    const int B = 256;
    auto blk = [](long long n, int b) { return (int)((n + b - 1) / b); };

    // --- normalization ---
    k_deg_init<<<blk(NN, B), B, 0, stream>>>(dis);
    k_deg_acc <<<blk(EE, B), B, 0, stream>>>(erow, ecol, dis);
    k_dis     <<<blk(NN, B), B, 0, stream>>>(dis);
    k_norm    <<<blk(EE, B), B, 0, stream>>>(erow, ecol, dis, norm);

    // --- CSR build (count -> scan -> fill -> sort) ---
    k_cnt_init <<<blk(NN, B), B, 0, stream>>>(csr_pos);            // reuse as cnt
    k_cnt      <<<blk(EE, B), B, 0, stream>>>(erow, ecol, csr_pos);
    k_scan     <<<1, SCAN_T, 0, stream>>>(csr_pos, csr_off, csr_pos);
    k_fill     <<<blk(EE, B), B, 0, stream>>>(erow, ecol, csr_pos, csr_eid);
    k_sortlists<<<blk(NN, B), B, 0, stream>>>(csr_off, csr_eid);

    // --- shared prep ---
    k_embT_init<<<blk((long long)EMBT_COLS * EMBT_LD, B), B, 0, stream>>>(embT);
    k_cvt_W    <<<blk(F_CAT * F_OUT, B), B, 0, stream>>>(W, Wt);

    // --- per-input: propagate (gather, no atomics), transform ---
    const float* feats[2] = { feat, feat_a };
    float* embOut[2] = { out_emb, emb_a };
    for (int which = 0; which < 2; ++which) {
        k_copy_hop0<<<blk((long long)NN * (F_IN / 4), B), B, 0, stream>>>(feats[which], xcat);
        for (int h = 1; h <= KHOPS; ++h) {
            k_gather<<<blk((long long)NN * (F_IN / 4), B), B, 0, stream>>>(
                csr_off, csr_eid, erow, norm, dis, xcat, h);
        }
        k_cvt_bf<<<blk((long long)NN * F_CAT / 8, B), B, 0, stream>>>(
            xcat, xbf, (long long)NN * F_CAT / 8);
        k_gemm_xW<<<blk(M_TILES, 8), 256, 0, stream>>>(xbf, Wt, embOut[which], embT, which * 64);
    }

    // --- readout GEMM over the 400MB mask (single pass, async double-buffer) ---
    k_gemm_mask<<<M_TILES, 288, 0, stream>>>(mask, embT, vsum);

    // --- finalize readout + discriminator ---
    k_readout<<<blk(NN, B), B, 0, stream>>>(vsum, g, ga);
    k_disc   <<<blk(NN, B), B, 0, stream>>>(out_emb, emb_a, g, ga, Wb, bsc, out_ret, out_ra);
}
